// QuantizedLinear_59768764891860
// MI455X (gfx1250) — 8-bit matrix operand order repaired and checked by reading, compiled, not run on hardware
//
#include <hip/hip_runtime.h>
#include <stdint.h>

#define M_DIM  8192
#define K_DIM  4096
#define N_DIM  4096
#define NBLK   32
#define BLK_K  128

#define TILE_M 128
#define TILE_N 128
#define LDSS   144

typedef int   v8i __attribute__((ext_vector_type(8)));
typedef float v8f __attribute__((ext_vector_type(8)));
typedef float v4f __attribute__((ext_vector_type(4)));
typedef int   i32x4 __attribute__((ext_vector_type(4)));
typedef int   i32x2 __attribute__((ext_vector_type(2)));

#if defined(__has_builtin)
#  if __has_builtin(__builtin_amdgcn_global_load_async_to_lds_b128)
#    define USE_ASYNC_LDS 1
#  endif
#endif
#ifndef USE_ASYNC_LDS
#  define USE_ASYNC_LDS 0
#endif

#if USE_ASYNC_LDS
typedef __attribute__((address_space(1))) i32x4 gi32x4;
typedef __attribute__((address_space(3))) i32x4 li32x4;

__device__ __forceinline__ void async_cp128(const void* g, void* l) {
  __builtin_amdgcn_global_load_async_to_lds_b128((gi32x4*)g, (li32x4*)l, 0, 0);
}
__device__ __forceinline__ void wait_async() {
#  if __has_builtin(__builtin_amdgcn_s_wait_asynccnt)
  __builtin_amdgcn_s_wait_asynccnt(0);
#  else
  asm volatile("s_wait_asynccnt 0x0" ::: "memory");
#  endif
}
#endif

__device__ __forceinline__ v8i pack8(int4 a, int4 b) {
  v8i r;
  r[0] = a.x; r[1] = a.y; r[2] = a.z; r[3] = a.w;
  r[4] = b.x; r[5] = b.y; r[6] = b.z; r[7] = b.w;
  return r;
}

__global__ __launch_bounds__(256) void quant_rows(const float* __restrict__ x,
                                                  int8_t* __restrict__ xq,
                                                  float* __restrict__ xscale) {
  const int m    = blockIdx.x;
  const int t    = threadIdx.x;
  const int lane = t & 31;
  const int wv   = t >> 5;

  const int g  = t >> 2;
  const int c  = t & 3;
  const int hf = c >> 1;
  const int j0 = (c & 1) * 2;

  const float* row = x + (size_t)m * K_DIM;
  const v4f* s0 = (const v4f*)(row + g * 64 + j0 * 16 + hf * 8);
  const v4f* s1 = (const v4f*)(row + g * 64 + (j0 + 1) * 16 + hf * 8);
  v4f v0 = __builtin_nontemporal_load(&s0[0]);
  v4f v1 = __builtin_nontemporal_load(&s0[1]);
  v4f v2 = __builtin_nontemporal_load(&s1[0]);
  v4f v3 = __builtin_nontemporal_load(&s1[1]);

  float amax = 0.0f;
  amax = fmaxf(amax, fabsf(v0.x)); amax = fmaxf(amax, fabsf(v0.y));
  amax = fmaxf(amax, fabsf(v0.z)); amax = fmaxf(amax, fabsf(v0.w));
  amax = fmaxf(amax, fabsf(v1.x)); amax = fmaxf(amax, fabsf(v1.y));
  amax = fmaxf(amax, fabsf(v1.z)); amax = fmaxf(amax, fabsf(v1.w));
  amax = fmaxf(amax, fabsf(v2.x)); amax = fmaxf(amax, fabsf(v2.y));
  amax = fmaxf(amax, fabsf(v2.z)); amax = fmaxf(amax, fabsf(v2.w));
  amax = fmaxf(amax, fabsf(v3.x)); amax = fmaxf(amax, fabsf(v3.y));
  amax = fmaxf(amax, fabsf(v3.z)); amax = fmaxf(amax, fabsf(v3.w));

  #pragma unroll
  for (int off = 16; off > 0; off >>= 1)
    amax = fmaxf(amax, __shfl_xor(amax, off, 32));

  __shared__ float red[8];
  __shared__ float s_scale;
  if (lane == 0) red[wv] = amax;
  __syncthreads();
  if (t == 0) {
    float a = red[0];
    #pragma unroll
    for (int i = 1; i < 8; ++i) a = fmaxf(a, red[i]);
    float sc = fmaxf(a, 1e-6f) * (1.0f / 127.0f);
    s_scale = sc;
    xscale[m] = sc;
  }
  __syncthreads();

  const float inv = 1.0f / s_scale;

  auto q8 = [&](float f) -> int {
    int q = __float2int_rn(f * inv);
    q = q < -128 ? -128 : q;
    q = q >  127 ?  127 : q;
    return q & 255;
  };

  int p0 = q8(v0.x) | (q8(v0.y) << 8) | (q8(v0.z) << 16) | (q8(v0.w) << 24);
  int p1 = q8(v1.x) | (q8(v1.y) << 8) | (q8(v1.z) << 16) | (q8(v1.w) << 24);
  int p2 = q8(v2.x) | (q8(v2.y) << 8) | (q8(v2.z) << 16) | (q8(v2.w) << 24);
  int p3 = q8(v3.x) | (q8(v3.y) << 8) | (q8(v3.z) << 16) | (q8(v3.w) << 24);

  ((int4*)(xq + (size_t)m * K_DIM))[t] = make_int4(p0, p1, p2, p3);
}

__global__ __launch_bounds__(256) void wmma_gemm(const int8_t* __restrict__ xq,
                                                 const float*  __restrict__ xscale,
                                                 const int8_t* __restrict__ wq,
                                                 const float*  __restrict__ wscale,
                                                 const float*  __restrict__ bias,
                                                 float*        __restrict__ out) {
  __shared__ int8_t lsA[2][TILE_M * LDSS];
  __shared__ int8_t lsB[2][TILE_N * LDSS];

  const int t    = threadIdx.x;
  const int lane = t & 31;
  const int wave = t >> 5;
  const int half = lane >> 4;
  const int lrow = lane & 15;
  const int wm   = wave >> 2;
  const int wn   = wave & 3;

  const int m0 = blockIdx.y * TILE_M;
  const int n0 = blockIdx.x * TILE_N;

  const int srow = t >> 1;
  const int soff = (t & 1) * 64;
  const int8_t* gA = xq + (size_t)(m0 + srow) * K_DIM + soff;
  const int8_t* gB = wq + (size_t)(n0 + srow) * K_DIM + soff;
  int8_t* dA0 = &lsA[0][srow * LDSS + soff];
  int8_t* dB0 = &lsB[0][srow * LDSS + soff];
  int8_t* dA1 = &lsA[1][srow * LDSS + soff];
  int8_t* dB1 = &lsB[1][srow * LDSS + soff];

#if USE_ASYNC_LDS
  auto stage = [&](int kofs, int buf) {
    const int8_t* sa = gA + kofs;
    const int8_t* sb = gB + kofs;
    int8_t* da = buf ? dA1 : dA0;
    int8_t* db = buf ? dB1 : dB0;
    #pragma unroll
    for (int j = 0; j < 4; ++j) {
      async_cp128(sa + 16 * j, da + 16 * j);
      async_cp128(sb + 16 * j, db + 16 * j);
    }
  };
  stage(0, 0);
  wait_async();
  __syncthreads();
#else
  int4 ra[4], rb[4];
  auto ld = [&](int kofs) {
    #pragma unroll
    for (int j = 0; j < 4; ++j) {
      ra[j] = *(const int4*)(gA + kofs + j * 16);
      rb[j] = *(const int4*)(gB + kofs + j * 16);
    }
  };
  auto st = [&](int buf) {
    int8_t* da = buf ? dA1 : dA0;
    int8_t* db = buf ? dB1 : dB0;
    #pragma unroll
    for (int j = 0; j < 4; ++j) {
      *(int4*)(da + 16 * j) = ra[j];
      *(int4*)(db + 16 * j) = rb[j];
    }
  };
  ld(0);
  st(0);
  __syncthreads();
#endif

  v8f facc[8];
  #pragma unroll
  for (int i = 0; i < 8; ++i)
    #pragma unroll
    for (int e = 0; e < 8; ++e) facc[i][e] = 0.0f;

  const int colbase = n0 + wn * 32 + lrow;

  for (int b = 0; b < NBLK; ++b) {
    const int cur = b & 1;

#if USE_ASYNC_LDS
    if (b + 1 < NBLK) stage((b + 1) * BLK_K, cur ^ 1);
#else
    if (b + 1 < NBLK) ld((b + 1) * BLK_K);
#endif

    const int8_t* sA = &lsA[cur][(wm * 64 + lrow) * LDSS + half * 32];
    const int8_t* sB = &lsB[cur][(wn * 32 + lrow) * LDSS + 8 * half];

    #pragma unroll
    for (int ns = 0; ns < 2; ++ns) {
      v8i iacc[4];
      #pragma unroll
      for (int tI = 0; tI < 4; ++tI)
        #pragma unroll
        for (int e = 0; e < 8; ++e) iacc[tI][e] = 0;

      #pragma unroll
      for (int h = 0; h < 2; ++h) {
        const int8_t* pb = sB + ns * 16 * LDSS + h * 64;
        v8i bf; { i32x2 b0 = *(const i32x2*)(pb), b1 = *(const i32x2*)(pb + 16), b2 = *(const i32x2*)(pb + 32), b3 = *(const i32x2*)(pb + 48); bf[0] = b0.x; bf[1] = b0.y; bf[2] = b1.x; bf[3] = b1.y; bf[4] = b2.x; bf[5] = b2.y; bf[6] = b3.x; bf[7] = b3.y; }

        #pragma unroll
        for (int tI = 0; tI < 4; ++tI) {
          const int8_t* pa = sA + tI * 16 * LDSS + h * 64;
          v8i af = pack8(*(const int4*)(pa), *(const int4*)(pa + 16));
          iacc[tI] = __builtin_amdgcn_wmma_i32_16x16x64_iu8(
               true, af,  true, bf, iacc[tI],
               false,  false);
        }
      }

      const float ws = wscale[(size_t)b * N_DIM + colbase + ns * 16];
      #pragma unroll
      for (int tI = 0; tI < 4; ++tI)
        #pragma unroll
        for (int e = 0; e < 8; ++e)
          facc[ns * 4 + tI][e] = fmaf((float)iacc[tI][e], ws, facc[ns * 4 + tI][e]);
    }

#if USE_ASYNC_LDS
    wait_async();
#else
    if (b + 1 < NBLK) st(cur ^ 1);
#endif
    __syncthreads();
  }

  #pragma unroll
  for (int ns = 0; ns < 2; ++ns) {
    const int n  = colbase + ns * 16;
    const float bv = bias[n];
    #pragma unroll
    for (int tI = 0; tI < 4; ++tI) {
      const int mb = m0 + wm * 64 + tI * 16 + half * 8;
      v8f xs = *(const v8f*)(&xscale[mb]);
      #pragma unroll
      for (int e = 0; e < 8; ++e) {
        float r = fmaf(facc[ns * 4 + tI][e], xs[e], bv);
        __builtin_nontemporal_store(r, &out[(size_t)(mb + e) * N_DIM + n]);
      }
    }
  }
}

extern "C" void kernel_launch(void* const* d_in, const int* in_sizes, int n_in,
                              void* d_out, int out_size, void* d_ws, size_t ws_size,
                              hipStream_t stream) {
  (void)in_sizes; (void)n_in; (void)out_size; (void)ws_size;

  const float*  x      = (const float*)d_in[0];
  const int8_t* wq     = (const int8_t*)d_in[1];
  const float*  wscale = (const float*)d_in[2];
  const float*  bias   = (const float*)d_in[3];
  float*        out    = (float*)d_out;

  int8_t* xq     = (int8_t*)d_ws;
  float*  xscale = (float*)((char*)d_ws + (size_t)M_DIM * K_DIM);

  quant_rows<<<M_DIM, 256, 0, stream>>>(x, xq, xscale);

  dim3 grid(N_DIM / TILE_N, M_DIM / TILE_M);
  wmma_gemm<<<grid, 256, 0, stream>>>(xq, xscale, wq, wscale, bias, out);
}
